// NNN_31791347925814
// MI455X (gfx1250) — matrix operand order repaired and checked by reading, not compiled, not run on hardware
//
#include <hip/hip_runtime.h>

typedef _Float16 v16h __attribute__((ext_vector_type(16)));
typedef _Float16 v8h  __attribute__((ext_vector_type(8)));
typedef float    v8f  __attribute__((ext_vector_type(8)));
typedef float    v4f  __attribute__((ext_vector_type(4)));

#define NSEQ (256 * 1024)   // stage-1 independent sequences
#define S2   1024           // stage-2 sequence length
#define B2   256            // batch
#define BT   16             // batch tile per workgroup (stage 2)
#define AST  136            // hbuf row stride in halves (272B: 16B aligned, conflict-skewed)
#define FST  132            // hfin row stride in floats

__device__ __forceinline__ float fsigm(float x) {
  // sigmoid(x) = 1 / (1 + 2^(-x*log2(e)))
  return __builtin_amdgcn_rcpf(1.0f + __builtin_amdgcn_exp2f(-1.44269504088896f * x));
}
__device__ __forceinline__ float ftanh_(float x) {
  // tanh(x) = 2*sigmoid(2x) - 1
  return 2.0f * __builtin_amdgcn_rcpf(1.0f + __builtin_amdgcn_exp2f(-2.88539008177793f * x)) - 1.0f;
}

// ---------------------------------------------------------------------------
// Stage 1: 262144 independent LSTMs, T=16, I=1, H=8. One thread per sequence.
// Writes final hidden state as f16 (A-operand-ready for stage 2 WMMA).
// ---------------------------------------------------------------------------
__global__ __launch_bounds__(256) void lstm1_kernel(
    const float* __restrict__ x, const float* __restrict__ W_ih1,
    const float* __restrict__ W_hh1, const float* __restrict__ b_ih1,
    const float* __restrict__ b_hh1, _Float16* __restrict__ h1out)
{
  __shared__ __align__(16) float xS[256 * 17];   // 17-stride: bank-conflict-free
  __shared__ __align__(16) float whS[32 * 8];
  __shared__ float wiS[32];
  __shared__ float bcS[32];

  const int tid = threadIdx.x;
  const int q0  = blockIdx.x * 256;

  // coalesced stage of this block's 256 sequences (16 floats each)
  for (int i = tid; i < 256 * 16; i += 256)
    xS[(i >> 4) * 17 + (i & 15)] = x[(size_t)q0 * 16 + i];
  if (tid < 32) { wiS[tid] = W_ih1[tid]; bcS[tid] = b_ih1[tid] + b_hh1[tid]; }
  if (tid < 256) whS[tid] = W_hh1[tid];
  __syncthreads();

  float h[8], c[8];
#pragma unroll
  for (int j = 0; j < 8; ++j) { h[j] = 0.0f; c[j] = 0.0f; }

  const float* xrow = &xS[tid * 17];
  for (int t = 0; t < 16; ++t) {
    const float xt = xrow[t];
    float ga[32];
#pragma unroll
    for (int gt = 0; gt < 32; ++gt) {
      const v4f* w = (const v4f*)&whS[gt * 8];
      const v4f w0 = w[0], w1 = w[1];
      float acc = bcS[gt] + wiS[gt] * xt;
      acc += w0.x * h[0] + w0.y * h[1] + w0.z * h[2] + w0.w * h[3];
      acc += w1.x * h[4] + w1.y * h[5] + w1.z * h[6] + w1.w * h[7];
      ga[gt] = acc;
    }
#pragma unroll
    for (int j = 0; j < 8; ++j) {
      const float ii = fsigm(ga[j]);
      const float ff = fsigm(ga[8 + j]);
      const float g2 = ftanh_(ga[16 + j]);
      const float oo = fsigm(ga[24 + j]);
      c[j] = ff * c[j] + ii * g2;
      h[j] = oo * ftanh_(c[j]);
    }
  }

  v8h hv;
#pragma unroll
  for (int j = 0; j < 8; ++j) hv[j] = (_Float16)h[j];
  *(v8h*)(h1out + (size_t)(q0 + tid) * 8) = hv;
}

// ---------------------------------------------------------------------------
// Stage 2: recurrent GEMM LSTM (H=128, I=8) via V_WMMA_F32_16X16X32_F16.
// 16 workgroups x 256 threads (8 waves). Wave w owns hidden cols [16w,16w+16)
// and all four gates for them (no cross-lane gather in the cell update).
// W_hh2/W_ih2 live in VGPRs as packed B operands; h double-buffers in LDS.
// Fused dense head at the end.
// ---------------------------------------------------------------------------
__global__ __launch_bounds__(256) void lstm2_kernel(
    const _Float16* __restrict__ h1, const float* __restrict__ W_ih2,
    const float* __restrict__ W_hh2, const float* __restrict__ b_ih2,
    const float* __restrict__ b_hh2, const float* __restrict__ W1,
    const float* __restrict__ b1, const float* __restrict__ W2,
    const float* __restrict__ b2, float* __restrict__ out)
{
  __shared__ __align__(16) _Float16 hbuf[2][16 * AST]; // f16 h state, double buffered
  __shared__ float hfin[16 * FST];                      // f32 final h for the head
  __shared__ float oacc[16];

  const int tid = threadIdx.x;
  const int wv  = tid >> 5;          // wave id: hidden group [16wv, 16wv+16)
  const int ln  = tid & 15;          // lane%16: A-row / B,C,D-column
  const int lg  = (tid >> 4) & 1;    // lane group (K-half selector)
  const int r0  = blockIdx.x * BT;   // first batch row of this workgroup

  for (int i = tid; i < 16 * AST; i += 256) hbuf[0][i] = (_Float16)0.0f;
  if (tid < 16) oacc[tid] = 0.0f;

  // ---- preload B operands (persistent in VGPRs) --------------------------
  // B[k][n] = W[col_n][k]; lane holds column n=ln, halves cover K=32c+16*lg+h
  v16h Bh[4][4];  // [gate][K-chunk] recurrent weights
  v16h Bx[4];     // input weights, K zero-padded 8->32
  float bias[4];
#pragma unroll
  for (int q = 0; q < 4; ++q) {
    const int coln = q * 128 + wv * 16 + ln;
    bias[q] = b_ih2[coln] + b_hh2[coln];
#pragma unroll
    for (int cc = 0; cc < 4; ++cc) {
      const float* wp = W_hh2 + coln * 128 + cc * 32 + lg * 16;
      v16h bb;
#pragma unroll
      for (int k = 0; k < 16; ++k) bb[k] = (_Float16)wp[k];
      Bh[q][cc] = bb;
    }
    v16h bx;
#pragma unroll
    for (int k = 0; k < 16; ++k) bx[k] = (_Float16)0.0f;
    if (lg == 0) {
#pragma unroll
      for (int k = 0; k < 8; ++k) bx[k] = (_Float16)W_ih2[coln * 8 + k];
    }
    Bx[q] = bx;
  }

  v8f cst;
#pragma unroll
  for (int e = 0; e < 8; ++e) cst[e] = 0.0f;

  v8h zero8;
#pragma unroll
  for (int k = 0; k < 8; ++k) zero8[k] = (_Float16)0.0f;

  // software-pipelined x-part A operand (one step ahead)
  v8h xa_next = zero8;
  if (lg == 0) xa_next = *(const v8h*)(h1 + ((size_t)(r0 + ln) * S2 + 0) * 8);

  __syncthreads();

  for (int t = 0; t < S2; ++t) {
    const _Float16* hb = &hbuf[t & 1][0];
    _Float16*       hn = &hbuf[(t + 1) & 1][0];

    const v8h xa = xa_next;
    if (lg == 0 && (t + 1) < S2)
      xa_next = *(const v8h*)(h1 + ((size_t)(r0 + ln) * S2 + (t + 1)) * 8);

    // A operands: row = ln; halves 0..15 -> K=32c+16lg+s (contiguous)
    const v16h Ax = __builtin_shufflevector(xa, zero8,
        0,1,2,3,4,5,6,7,8,9,10,11,12,13,14,15);
    v16h Ar[4];
#pragma unroll
    for (int cc = 0; cc < 4; ++cc) {
      const v8h lo = *(const v8h*)(hb + ln * AST + cc * 32 + lg * 16);
      const v8h hi = *(const v8h*)(hb + ln * AST + cc * 32 + lg * 16 + 8);
      Ar[cc] = __builtin_shufflevector(lo, hi,
          0,1,2,3,4,5,6,7,8,9,10,11,12,13,14,15);
    }

    v8f acc[4];
#pragma unroll
    for (int q = 0; q < 4; ++q) {
      v8f a;
#pragma unroll
      for (int e = 0; e < 8; ++e) a[e] = bias[q];
      a = __builtin_amdgcn_wmma_f32_16x16x32_f16(false, Ax, false, Bx[q],
                                                 (short)0, a, false, false);
#pragma unroll
      for (int cc = 0; cc < 4; ++cc)
        a = __builtin_amdgcn_wmma_f32_16x16x32_f16(false, Ar[cc], false,
                                                   Bh[q][cc], (short)0, a,
                                                   false, false);
      acc[q] = a;
    }

    // cell update: element e is (row m = e+8*lg, hidden col wv*16+ln)
#pragma unroll
    for (int e = 0; e < 8; ++e) {
      const float ii = fsigm(acc[0][e]);
      const float ff = fsigm(acc[1][e]);
      const float g2 = ftanh_(acc[2][e]);
      const float oo = fsigm(acc[3][e]);
      const float cv = ff * cst[e] + ii * g2;
      cst[e] = cv;
      const float hv = oo * ftanh_(cv);
      hn[(e + 8 * lg) * AST + wv * 16 + ln] = (_Float16)hv;
      if (t == S2 - 1) hfin[(e + 8 * lg) * FST + wv * 16 + ln] = hv;
    }
    __syncthreads();
  }

  // ---- fused dense head: out = (h@W1^T + b1)@W2^T + b2 -------------------
  const int m  = tid >> 4;   // batch row in tile
  const int jq = tid & 15;   // handles j = 4*jq .. 4*jq+3
  float part = 0.0f;
#pragma unroll
  for (int jj = 0; jj < 4; ++jj) {
    const int j = jq * 4 + jj;
    float s = b1[j];
    for (int k = 0; k < 128; ++k) s += hfin[m * FST + k] * W1[j * 128 + k];
    part += s * W2[j];
  }
  atomicAdd(&oacc[m], part);
  __syncthreads();
  if (tid < 16) out[r0 + tid] = oacc[tid] + b2[0];
}

// ---------------------------------------------------------------------------
extern "C" void kernel_launch(void* const* d_in, const int* in_sizes, int n_in,
                              void* d_out, int out_size, void* d_ws, size_t ws_size,
                              hipStream_t stream) {
  const float* x     = (const float*)d_in[0];
  // d_in[1] = 'data' scalar, unused
  const float* W_ih1 = (const float*)d_in[2];
  const float* W_hh1 = (const float*)d_in[3];
  const float* b_ih1 = (const float*)d_in[4];
  const float* b_hh1 = (const float*)d_in[5];
  const float* W_ih2 = (const float*)d_in[6];
  const float* W_hh2 = (const float*)d_in[7];
  const float* b_ih2 = (const float*)d_in[8];
  const float* b_hh2 = (const float*)d_in[9];
  const float* W1    = (const float*)d_in[10];
  const float* b1    = (const float*)d_in[11];
  const float* W2    = (const float*)d_in[12];
  const float* b2    = (const float*)d_in[13];
  float* out = (float*)d_out;

  _Float16* h1 = (_Float16*)d_ws;  // [256*1024, 8] f16 stage-1 output (4 MB)

  lstm1_kernel<<<NSEQ / 256, 256, 0, stream>>>(x, W_ih1, W_hh1, b_ih1, b_hh1, h1);
  lstm2_kernel<<<B2 / BT, 256, 0, stream>>>(h1, W_ih2, W_hh2, b_ih2, b_hh2,
                                            W1, b1, W2, b2, out);
}
